// BGRL_74680891343655
// MI455X (gfx1250) — matrix operand order repaired and checked by reading, not compiled, not run on hardware
//
#include <hip/hip_runtime.h>

// ---------- types ----------
typedef __attribute__((ext_vector_type(16))) __bf16  v16bf;
typedef __attribute__((ext_vector_type(8)))  float   v8f;
typedef __attribute__((ext_vector_type(4)))  unsigned int v4u;
typedef __attribute__((ext_vector_type(2)))  unsigned int v2u;

union Frag { v16bf v; v4u q[2]; };

// gfx1250 async global->LDS path (ASYNCcnt-tracked, bypasses VGPRs)
// Probe-derived builtin signature:
//   void __builtin_amdgcn_global_load_async_to_lds_b128(int4 __device__* src,
//                                                       int4 __shared__* dst,
//                                                       Ii offset, Ii cpol)
#if __has_builtin(__builtin_amdgcn_global_load_async_to_lds_b128)
#define USE_ASYNC_LDS 1
typedef int v4i_ __attribute__((__vector_size__(4 * sizeof(int))));
typedef __attribute__((address_space(1))) v4i_ v4i_g;   // AS1 int4
typedef __attribute__((address_space(3))) v4i_ v4i_l;   // AS3 int4
#else
#define USE_ASYNC_LDS 0
#endif

__device__ __forceinline__ void async_wait0() {
#if __has_builtin(__builtin_amdgcn_s_wait_asynccnt)
    __builtin_amdgcn_s_wait_asynccnt(0);
#else
    asm volatile("s_wait_asynccnt 0x0" ::: "memory");
#endif
}

__device__ __forceinline__ unsigned short f2bf(float f) {
    unsigned int u = __float_as_uint(f);
    unsigned int r = (u + 0x7FFFu + ((u >> 16) & 1u)) >> 16;   // RNE
    return (unsigned short)r;
}

__device__ __forceinline__ void atomAddF(float* p, float v) {
    __hip_atomic_fetch_add(p, v, __ATOMIC_RELAXED, __HIP_MEMORY_SCOPE_AGENT);
}

// ---------- utility kernels ----------
__global__ void zerof_k(float* p, long long n) {
    long long i = (long long)blockIdx.x * blockDim.x + threadIdx.x;
    long long s = (long long)gridDim.x * blockDim.x;
    for (; i < n; i += s) p[i] = 0.0f;
}

__global__ void degrees_k(const int* __restrict__ src, const int* __restrict__ dst,
                          float* __restrict__ dout, float* __restrict__ din, int E) {
    int e = blockIdx.x * blockDim.x + threadIdx.x;
    if (e < E) {
        atomAddF(&dout[src[e]], 1.0f);
        atomAddF(&din[dst[e]], 1.0f);
    }
}

// feat_bf16 = bf16(features * rsqrt(deg_out[row]))   (fuses norm_src into A)
__global__ void cvt_feat_k(const float* __restrict__ feat, const float* __restrict__ deg,
                           unsigned short* __restrict__ out, long long total) {
    long long i = (long long)blockIdx.x * blockDim.x + threadIdx.x;
    long long s = (long long)gridDim.x * blockDim.x;
    for (; i < total; i += s) {
        long long row = i >> 9;                       // F = 512
        out[i] = f2bf(feat[i] * __frsqrt_rn(deg[row]));
    }
}

// Pack row-major W[K x N] (f32) into WMMA bf16 B-fragment dword layout:
// out[((ntile*kchunks + kt)*32 + lane)*8 + r] = { B[k][n], B[k+1][n] }
// n = ntile*16 + (lane&15);  k = kt*32 + (lane>>4)*8 + (r>>2)*16 + 2*(r&3)
__global__ void pack_B_k(const float* __restrict__ W, unsigned int* __restrict__ out,
                         int K, int N) {
    int kchunks = K >> 5;
    int total = (N >> 4) * kchunks * 256;
    for (int idx = blockIdx.x * blockDim.x + threadIdx.x; idx < total;
         idx += gridDim.x * blockDim.x) {
        int r     = idx & 7;
        int lane  = (idx >> 3) & 31;
        int chunk = idx >> 8;
        int kt    = chunk % kchunks;
        int ntile = chunk / kchunks;
        int n = (ntile << 4) + (lane & 15);
        int k = (kt << 5) + ((lane >> 4) << 3) + ((r >> 2) << 4) + ((r & 3) << 1);
        unsigned int lo = f2bf(W[(size_t)k * N + n]);
        unsigned int hi = f2bf(W[(size_t)(k + 1) * N + n]);
        out[idx] = lo | (hi << 16);
    }
}

// ---------- GEMM1: h[M x 256] = featb[M x 512](bf16) @ Wc_packed ----------
// grid (M/16, 2), block 256 (8 waves). Wave w -> N-tile (blockIdx.y*8 + w).
#define LDA1 520   // 512 + 8 shorts padding (bank-conflict stagger); 1040B row = 65*16
__global__ __launch_bounds__(256) void gemm1_wmma_k(
    const unsigned short* __restrict__ A,   // [M x 512] bf16, pre-normed
    const unsigned int*   __restrict__ Bp,  // packed B fragments
    float* __restrict__ Cout, int M) {
    __shared__ unsigned short lA[16 * LDA1];

    const int rowbase = blockIdx.x * 16;
#if USE_ASYNC_LDS
    {   // async-stage 16x512 bf16 A tile: 1024 x b128, 4 per thread
        const unsigned short* gsrc = A + (size_t)rowbase * 512;
        for (int i = threadIdx.x; i < 1024; i += 256) {
            int r = i >> 6, c = i & 63;   // row, 16B chunk within row
            __builtin_amdgcn_global_load_async_to_lds_b128(
                (v4i_g*)(gsrc + r * 512 + c * 8),
                (v4i_l*)(lA + r * LDA1 + c * 8), 0, 0);
        }
        async_wait0();
    }
#else
    {   // fallback: stage through VGPRs
        const v2u* srcp = (const v2u*)(A + (size_t)rowbase * 512);
        for (int i = threadIdx.x; i < 16 * 128; i += 256) {
            int r = i >> 7, c = i & 127;
            *(v2u*)(lA + r * LDA1 + c * 4) = srcp[i];
        }
    }
#endif
    __syncthreads();

    const int wave  = threadIdx.x >> 5;
    const int lane  = threadIdx.x & 31;
    const int ntile = blockIdx.y * 8 + wave;          // 0..15
    const int row   = lane & 15;
    const int half  = lane >> 4;

    v8f acc = {};
    const unsigned int* bptr = Bp + (size_t)ntile * 16 * 256 + lane * 8;
    #pragma unroll 4
    for (int kt = 0; kt < 16; ++kt) {
        Frag a, b;
        // A frag: regs0-3 = K half*8+{0..7}; regs4-7 = +16  -> two ds_load_b128
        const v4u* ap = (const v4u*)(lA + row * LDA1 + kt * 32 + half * 8);
        a.q[0] = ap[0];
        a.q[1] = ap[2];
        // B frag: 8 contiguous dwords per lane -> two global_load_b128
        const v4u* bp4 = (const v4u*)bptr;
        b.q[0] = bp4[0];
        b.q[1] = bp4[1];
        bptr += 256;
        acc = __builtin_amdgcn_wmma_f32_16x16x32_bf16(
                  false, a.v, false, b.v, (short)0, acc, false, false);
    }

    const int col   = ntile * 16 + (lane & 15);
    const int mbase = rowbase + half * 8;
    #pragma unroll
    for (int r = 0; r < 8; ++r)
        Cout[(size_t)(mbase + r) * 256 + col] = acc[r];
}

// ---------- SpMM scatter: agg[dst] += h[src]; one block per edge ----------
__global__ __launch_bounds__(256) void scatter_k(
    const int* __restrict__ src, const int* __restrict__ dst,
    const float* __restrict__ h, float* __restrict__ agg, int E) {
    int e = blockIdx.x;
    if (e >= E) return;
    int s = src[e], d = dst[e];
    int t = threadIdx.x;                              // feature dim, H=256
    float v = h[(size_t)s * 256 + t];
    atomAddF(&agg[(size_t)d * 256 + t], v);
}

// x = relu(agg * rsqrt(deg_in[row]) + b_conv[col]) -> bf16
__global__ void relu_bias_cvt_k(const float* __restrict__ agg,
                                const float* __restrict__ deg_in,
                                const float* __restrict__ bias,
                                unsigned short* __restrict__ out, long long total) {
    long long i = (long long)blockIdx.x * blockDim.x + threadIdx.x;
    long long s = (long long)gridDim.x * blockDim.x;
    for (; i < total; i += s) {
        long long row = i >> 8;                       // H = 256
        int col = (int)(i & 255);
        float v = agg[i] * __frsqrt_rn(deg_in[row]) + bias[col];
        out[i] = f2bf(fmaxf(v, 0.0f));
    }
}

// ---------- GEMM2: logits[M x 64] = xb[M x 256](bf16) @ W2_packed ----------
// grid (M/16), block 128 (4 waves). Wave w -> N-tile w.
#define LDA2 264   // 256 + 8 shorts padding; 528B row = 33*16
__global__ __launch_bounds__(128) void gemm2_wmma_k(
    const unsigned short* __restrict__ A,   // [M x 256] bf16
    const unsigned int*   __restrict__ Bp,
    float* __restrict__ Cout, int M) {
    __shared__ unsigned short lA[16 * LDA2];

    const int rowbase = blockIdx.x * 16;
#if USE_ASYNC_LDS
    {   // async-stage 16x256 bf16 A tile: 512 x b128, 4 per thread
        const unsigned short* gsrc = A + (size_t)rowbase * 256;
        for (int i = threadIdx.x; i < 512; i += 128) {
            int r = i >> 5, c = i & 31;
            __builtin_amdgcn_global_load_async_to_lds_b128(
                (v4i_g*)(gsrc + r * 256 + c * 8),
                (v4i_l*)(lA + r * LDA2 + c * 8), 0, 0);
        }
        async_wait0();
    }
#else
    {
        const v2u* srcp = (const v2u*)(A + (size_t)rowbase * 256);
        for (int i = threadIdx.x; i < 16 * 64; i += 128) {
            int r = i >> 6, c = i & 63;
            *(v2u*)(lA + r * LDA2 + c * 4) = srcp[i];
        }
    }
#endif
    __syncthreads();

    const int wave = threadIdx.x >> 5;                // ntile 0..3
    const int lane = threadIdx.x & 31;
    const int row  = lane & 15;
    const int half = lane >> 4;

    v8f acc = {};
    const unsigned int* bptr = Bp + (size_t)wave * 8 * 256 + lane * 8;
    #pragma unroll
    for (int kt = 0; kt < 8; ++kt) {
        Frag a, b;
        const v4u* ap = (const v4u*)(lA + row * LDA2 + kt * 32 + half * 8);
        a.q[0] = ap[0];
        a.q[1] = ap[2];
        const v4u* bp4 = (const v4u*)bptr;
        b.q[0] = bp4[0];
        b.q[1] = bp4[1];
        bptr += 256;
        acc = __builtin_amdgcn_wmma_f32_16x16x32_bf16(
                  false, a.v, false, b.v, (short)0, acc, false, false);
    }

    const int col   = wave * 16 + (lane & 15);
    const int mbase = rowbase + half * 8;
    #pragma unroll
    for (int r = 0; r < 8; ++r)
        Cout[(size_t)(mbase + r) * 64 + col] = acc[r];
}

// ---------- fused bias + log_softmax, one wave32 per row (C = 64) ----------
__global__ __launch_bounds__(256) void logsoftmax_k(
    const float* __restrict__ logits, const float* __restrict__ b2,
    float* __restrict__ out, int M) {
    int row  = blockIdx.x * 8 + (threadIdx.x >> 5);
    int lane = threadIdx.x & 31;
    if (row >= M) return;
    float v0 = logits[(size_t)row * 64 + lane]      + b2[lane];
    float v1 = logits[(size_t)row * 64 + 32 + lane] + b2[32 + lane];
    float m = fmaxf(v0, v1);
    for (int o = 16; o; o >>= 1) m = fmaxf(m, __shfl_xor(m, o, 32));
    float s = __expf(v0 - m) + __expf(v1 - m);
    for (int o = 16; o; o >>= 1) s += __shfl_xor(s, o, 32);
    float l = __logf(s);
    out[(size_t)row * 64 + lane]      = v0 - m - l;
    out[(size_t)row * 64 + 32 + lane] = v1 - m - l;
}

// ---------- host ----------
extern "C" void kernel_launch(void* const* d_in, const int* in_sizes, int n_in,
                              void* d_out, int out_size, void* d_ws, size_t ws_size,
                              hipStream_t stream) {
    const float* features = (const float*)d_in[0];
    const float* W_conv   = (const float*)d_in[1];
    const float* b_conv   = (const float*)d_in[2];
    const float* W2       = (const float*)d_in[3];
    const float* b2       = (const float*)d_in[4];
    const int*   src      = (const int*)d_in[5];
    const int*   dst      = (const int*)d_in[6];

    const int F = 512, H = 256, C = 64;
    const int E = in_sizes[5];
    const int M = in_sizes[0] / F;                    // 50000 (divisible by 16)

    char* ws = (char*)d_ws;
    size_t off = 0;
    auto alloc = [&](size_t bytes) -> void* {
        void* p = ws + off;
        off = (off + bytes + 255) & ~(size_t)255;
        return p;
    };
    float*          deg_out = (float*)alloc((size_t)M * 4);
    float*          deg_in  = (float*)alloc((size_t)M * 4);
    float*          agg     = (float*)alloc((size_t)M * H * 4);
    unsigned short* featb   = (unsigned short*)alloc((size_t)M * F * 2);
    unsigned int*   Wcp     = (unsigned int*)alloc((size_t)F * H * 2);
    float*          h       = (float*)alloc((size_t)M * H * 4);
    unsigned short* xb      = (unsigned short*)alloc((size_t)M * H * 2);
    unsigned int*   W2p     = (unsigned int*)alloc((size_t)H * C * 2);
    float*          logits  = (float*)alloc((size_t)M * C * 4);

    zerof_k<<<1024, 256, 0, stream>>>(deg_out, M);
    zerof_k<<<1024, 256, 0, stream>>>(deg_in, M);
    zerof_k<<<4096, 256, 0, stream>>>(agg, (long long)M * H);

    degrees_k<<<(E + 255) / 256, 256, 0, stream>>>(src, dst, deg_out, deg_in, E);

    cvt_feat_k<<<8192, 256, 0, stream>>>(features, deg_out, featb, (long long)M * F);
    pack_B_k<<<256, 256, 0, stream>>>(W_conv, Wcp, F, H);

    gemm1_wmma_k<<<dim3(M / 16, 2), 256, 0, stream>>>(featb, Wcp, h, M);

    scatter_k<<<E, 256, 0, stream>>>(src, dst, h, agg, E);

    relu_bias_cvt_k<<<4096, 256, 0, stream>>>(agg, deg_in, b_conv, xb, (long long)M * H);
    pack_B_k<<<32, 256, 0, stream>>>(W2, W2p, H, C);

    gemm2_wmma_k<<<M / 16, 128, 0, stream>>>(xb, W2p, logits, M);

    logsoftmax_k<<<(M + 7) / 8, 256, 0, stream>>>(logits, b2, (float*)d_out, M);
}
